// threeModuleRNN_sparse_context_gating_89618787599002
// MI455X (gfx1250) — matrix operand order repaired and checked by reading, not compiled, not run on hardware
//
#include <hip/hip_runtime.h>
#include <hip/hip_bf16.h>

typedef __bf16 bf16;
typedef __attribute__((ext_vector_type(16))) __bf16 bf16x16;
typedef __attribute__((ext_vector_type(8)))  float   floatx8;
typedef int v4i_vec __attribute__((vector_size(16)));

#define BATCH 32
#define NMOD  1024
#define NTOT  3072
#define TSTEPS 512
#define TAUINV 0.1f
#define NWG_MAIN 96
#define NWG_TOTAL 101

// ---------------- WMMA helpers (CDNA5 16x16x32 bf16, wave32) ----------------

__device__ __forceinline__ floatx8 wmma_bf16(bf16x16 a, bf16x16 b, floatx8 c) {
  // v_wmma_f32_16x16x32_bf16: D = A(16x32) x B(32x16) + C
  return __builtin_amdgcn_wmma_f32_16x16x32_bf16(false, a, false, b, (short)0, c, false, false);
}

// A-matrix 16x32 bf16 fragment for this lane (ISA 7.12.2):
// lane l, row m = l&15, khalf = l>>4. Values: k = kb+khalf*8+{0..7} and kb+16+khalf*8+{0..7}
__device__ __forceinline__ bf16x16 load_a16(const bf16* row, int kb, int khalf) {
  bf16x16 v;
  const int k0 = kb + khalf * 8;
  *(float4*)&v         = *(const float4*)(row + k0);
  *(((float4*)&v) + 1) = *(const float4*)(row + k0 + 16);
  return v;
}

// B-matrix 32x16 bf16 fragment: lane l, col n = l&15 (caller selects row ptr = W[n]),
// values k = kb + (l>>4)*8 + {0..7} and kb + 16 + (l>>4)*8 + {0..7}
__device__ __forceinline__ bf16x16 load_b16(const bf16* wrow, int kb, int khalf) {
  bf16x16 v;
  const int k0 = kb + khalf * 8;
  *(float4*)&v         = *(const float4*)(wrow + k0);
  *(((float4*)&v) + 1) = *(const float4*)(wrow + k0 + 16);
  return v;
}

// ---------------- async global->LDS staging (CDNA5 ASYNCcnt path) ----------------

__device__ __forceinline__ void async_copy_b128(const bf16* gsrc, bf16* ldst) {
#if __has_builtin(__builtin_amdgcn_global_load_async_to_lds_b128)
  __builtin_amdgcn_global_load_async_to_lds_b128(
      (__attribute__((address_space(1))) v4i_vec*)(void*)gsrc,
      (__attribute__((address_space(3))) v4i_vec*)(void*)ldst, 0, 0);
#else
  *(uint4*)ldst = *(const uint4*)gsrc;
#endif
}

__device__ __forceinline__ void async_copy_wait() {
#if __has_builtin(__builtin_amdgcn_global_load_async_to_lds_b128)
#if __has_builtin(__builtin_amdgcn_s_wait_asynccnt)
  __builtin_amdgcn_s_wait_asynccnt(0);
#else
  asm volatile("s_wait_asynccnt 0x0" ::: "memory");
#endif
#endif
}

// ---------------- device-wide sense barrier ----------------

__device__ __forceinline__ void grid_barrier(int* cnt, int* gen, int nwg, int& lg) {
  __threadfence();          // release this thread's stores
  __syncthreads();
  if (threadIdx.x == 0) {
    ++lg;
    if (atomicAdd(cnt, 1) == nwg - 1) {
      atomicExch(cnt, 0);
      __threadfence();
      atomicExch(gen, lg);
    } else {
      while (__hip_atomic_load(gen, __ATOMIC_ACQUIRE, __HIP_MEMORY_SCOPE_AGENT) < lg)
        __builtin_amdgcn_s_sleep(1);
    }
    __threadfence();
  }
  __syncthreads();
}

// ---------------- shared memory union ----------------

struct MainSm {            // GEMM shard wg: 141,312 B
  bf16 W[32][NMOD];        // weight slice, row = local out col (B-operand friendly)
  bf16 th[BATCH][NMOD];    // staged tanh(h) module block
  float h[BATCH][32];      // persistent fp32 state slice
  bf16 Wih[32][32];        // module-1 input weights (K padded 16->32)
  float red[4][32][8];     // K-split reduction
};
struct FmSm {              // sparse projection wg: 107,520 B
  bf16 Wp[208][224];       // W_H_12 / W_H_23 zero-padded
  bf16 thg[BATCH][224];    // gathered th columns
};
struct RoSm {              // readout wg: 70,144 B
  bf16 Wr[32][NMOD];
  float part[32][4];
  float who[NMOD];
};
union SMem { MainSm m; FmSm f; RoSm r; };

// ---------------- setup kernels ----------------

__global__ void setup_misc(const float* __restrict__ h0,
                           const int* __restrict__ idN12_2, const int* __restrict__ idN23_3,
                           bf16* __restrict__ th, int* __restrict__ inv12,
                           int* __restrict__ inv23, int* __restrict__ bar) {
  const int tid = threadIdx.x;
  if (tid == 0) { bar[0] = 0; bar[1] = 0; }
  for (int i = tid; i < NMOD; i += 256) { inv12[i] = -1; inv23[i] = -1; }
  __syncthreads();
  for (int i = tid; i < 205; i += 256) { inv12[idN12_2[i]] = i; inv23[idN23_3[i]] = i; }
  for (int i = tid; i < BATCH * NTOT; i += 256) th[i] = (bf16)tanhf(h0[i]);
}

__global__ void setup_noise(const float* __restrict__ iWN, float* __restrict__ wn) {
  const int n = blockIdx.x * 256 + threadIdx.x;
  if (n >= NTOT) return;
  const float a  = expf(-1.0f / 10.0f);
  const float sq = sqrtf(10.0f);
  float prev = sq * iWN[(size_t)n * TSTEPS];
  wn[n] = 0.01f * prev;
  for (int t = 1; t < TSTEPS; ++t) {
    const float cur = sq * iWN[(size_t)n * TSTEPS + t];
    prev = cur + (prev - cur) * a;
    wn[(size_t)t * NTOT + n] = 0.01f * prev;
  }
}

__global__ void setup_x(const float* __restrict__ x, bf16* __restrict__ xbf) {
  const int t = blockIdx.x;
  for (int i = threadIdx.x; i < BATCH * 32; i += 256) {
    const int b = i >> 5, k = i & 31;
    xbf[((size_t)t * BATCH + b) * 32 + k] =
        (k < 16) ? (bf16)x[((size_t)b * 16 + k) * TSTEPS + t] : (bf16)0.0f;
  }
}

// ---------------- persistent RNN kernel ----------------

__global__ __launch_bounds__(256, 1)
void rnn_persistent(const float* __restrict__ h0, const float* __restrict__ context,
                    const float* __restrict__ W_ih, const float* __restrict__ W11,
                    const float* __restrict__ W22, const float* __restrict__ W33,
                    const float* __restrict__ W12, const float* __restrict__ W23,
                    const float* __restrict__ Who, const float* __restrict__ WHc2,
                    const float* __restrict__ Wr1, const float* __restrict__ Wr2,
                    const float* __restrict__ Wr3,
                    const int* __restrict__ idN12_1, const int* __restrict__ idN23_2,
                    bf16* __restrict__ th, const float* __restrict__ wn,
                    const bf16* __restrict__ xbf,
                    float* __restrict__ fm1, float* __restrict__ fm2,
                    const int* __restrict__ inv12, const int* __restrict__ inv23,
                    int* __restrict__ bar,
                    float* __restrict__ outH, float* __restrict__ outY,
                    float* __restrict__ outYR) {
  __shared__ SMem sm;
  const int wg   = blockIdx.x;
  const int tid  = threadIdx.x;
  const int lane = tid & 31;
  const int wv   = tid >> 5;
  const int halfL = lane >> 4;
  const int l15   = lane & 15;
  int lg = 0;
  int* bcnt = bar;
  int* bgen = bar + 1;

  if (wg < NWG_MAIN) {
    // -------- main GEMM shard: 32 output columns of one module --------
    const int mod = wg >> 5;                 // 0..2
    const int sl  = wg & 31;                 // slice within module
    const int nbase   = mod * NMOD + sl * 32;
    const int modbase = mod * NMOD;
    const float* Wsrc = (mod == 0) ? W11 : (mod == 1) ? W22 : W33;

    for (int i = tid; i < 32 * NMOD; i += 256) {
      const int n = i >> 10, k = i & 1023;
      sm.m.W[n][k] = (bf16)Wsrc[(size_t)(sl * 32 + n) * NMOD + k];
    }
    if (mod == 0) {
      for (int i = tid; i < 32 * 32; i += 256) {
        const int n = i >> 5, k = i & 31;
        sm.m.Wih[n][k] = (k < 16) ? (bf16)W_ih[(sl * 32 + n) * 16 + k] : (bf16)0.0f;
      }
    }
    for (int i = tid; i < BATCH * 32; i += 256) {
      const int b = i >> 5, j = i & 31;
      sm.m.h[b][j] = h0[(size_t)b * NTOT + nbase + j];
    }
    __syncthreads();

    const int tile  = wv & 3;     // (Mt,Nt) tile
    const int khalf = wv >> 2;    // K split half
    const int Mt = tile >> 1, Nt = tile & 1;
    const int n_l = Nt * 16 + l15;
    const int n_g = nbase + n_l;

    int jscat = -1;
    float cival = 0.0f;
    if (mod == 1) { jscat = inv12[n_g - NMOD]; cival = context[0] * WHc2[(n_g - NMOD) * 2]; }
    else if (mod == 2) { jscat = inv23[n_g - 2 * NMOD]; }

    for (int t = 0; t < TSTEPS; ++t) {
      // stage tanh(h) module block into LDS via the async ASYNCcnt path
      // (no VGPR round trip; LDS written directly by the async unit)
      for (int i = tid; i < BATCH * 128; i += 256) {
        const int b = i >> 7, q = i & 127;
        async_copy_b128(th + (size_t)b * NTOT + modbase + q * 8, &sm.m.th[b][q * 8]);
      }
      async_copy_wait();
      __syncthreads();

      floatx8 acc = {0, 0, 0, 0, 0, 0, 0, 0};
      const bf16* arow = &sm.m.th[Mt * 16 + l15][0];
      const bf16* brow = &sm.m.W[n_l][0];
      const int kend = khalf * 512 + 512;
      for (int kb = khalf * 512; kb < kend; kb += 32)
        acc = wmma_bf16(load_a16(arow, kb, halfL), load_b16(brow, kb, halfL), acc);

      if (mod == 0 && khalf == 0) {
        // + x_t @ W_ih.T  (K padded to 32)
        const bf16* xr = xbf + ((size_t)t * BATCH + (Mt * 16 + l15)) * 32;
        acc = wmma_bf16(load_a16(xr, 0, halfL), load_b16(&sm.m.Wih[n_l][0], 0, halfL), acc);
      }
      if (khalf == 1) {
        #pragma unroll
        for (int q = 0; q < 8; ++q) sm.m.red[tile][lane][q] = acc[q];
      }

      grid_barrier(bcnt, bgen, NWG_TOTAL, lg);   // B1: fromM/readout ready

      if (khalf == 0) {
        #pragma unroll
        for (int q = 0; q < 8; ++q) acc[q] += sm.m.red[tile][lane][q];
        const float wnv = wn[(size_t)t * NTOT + n_g];
        #pragma unroll
        for (int q = 0; q < 8; ++q) {
          const int b = Mt * 16 + halfL * 8 + q;   // C/D layout: M = r + 8*(lane>>4)
          float s = acc[q] + wnv;
          if (jscat >= 0)
            s += (mod == 1) ? fm1[b * 208 + jscat] * cival : fm2[b * 208 + jscat];
          const float hold = sm.m.h[b][n_l];
          const float hnew = hold + (s - hold) * TAUINV;
          const float thn  = tanhf(hnew);
          sm.m.h[b][n_l] = hnew;
          th[(size_t)b * NTOT + n_g] = (bf16)thn;
          outH[((size_t)b * NTOT + n_g) * TSTEPS + t] = thn;
        }
      }
      grid_barrier(bcnt, bgen, NWG_TOTAL, lg);   // B2: step complete
    }
  } else if (wg < NWG_MAIN + 2) {
    // -------- sparse cross projections: fromM1 / fromM2 (32x205 @ 205x205) --------
    const int which = wg - NWG_MAIN;
    const float* Wsp = which ? W23 : W12;
    const int*  gidx = which ? idN23_2 : idN12_1;
    float*      fout = which ? fm2 : fm1;
    const int thbase = which ? NMOD : 0;

    for (int i = tid; i < 208 * 224; i += 256) {
      const int jj = i / 224, k = i % 224;
      sm.f.Wp[jj][k] = (jj < 205 && k < 205) ? (bf16)Wsp[jj * 205 + k] : (bf16)0.0f;
    }
    __syncthreads();

    for (int t = 0; t < TSTEPS; ++t) {
      for (int i = tid; i < BATCH * 224; i += 256) {
        const int b = i / 224, k = i % 224;
        sm.f.thg[b][k] = (k < 205) ? th[(size_t)b * NTOT + thbase + gidx[k]] : (bf16)0.0f;
      }
      __syncthreads();
      for (int tile = wv; tile < 26; tile += 8) {     // 2 Mt x 13 Nt
        const int Mt = tile & 1, Ntl = tile >> 1;
        floatx8 acc = {0, 0, 0, 0, 0, 0, 0, 0};
        const bf16* arow = &sm.f.thg[Mt * 16 + l15][0];
        const bf16* brow = &sm.f.Wp[Ntl * 16 + l15][0];
        for (int kb = 0; kb < 224; kb += 32)
          acc = wmma_bf16(load_a16(arow, kb, halfL), load_b16(brow, kb, halfL), acc);
        const int jcol = Ntl * 16 + l15;
        if (jcol < 205) {
          #pragma unroll
          for (int q = 0; q < 8; ++q)
            fout[(Mt * 16 + halfL * 8 + q) * 208 + jcol] = acc[q];
        }
      }
      grid_barrier(bcnt, bgen, NWG_TOTAL, lg);   // B1
      grid_barrier(bcnt, bgen, NWG_TOTAL, lg);   // B2
    }
  } else {
    // -------- readouts: r_mod = tanh(th_mod @ W_ho_ran.T), plus y for mod==2 --------
    const int mod = wg - (NWG_MAIN + 2);
    const float* Wrp = (mod == 0) ? Wr1 : (mod == 1) ? Wr2 : Wr3;
    for (int i = tid; i < 32 * NMOD; i += 256) {
      const int rr = i >> 10, k = i & 1023;
      sm.r.Wr[rr][k] = (bf16)Wrp[rr * NMOD + k];
    }
    if (mod == 2)
      for (int i = tid; i < NMOD; i += 256) sm.r.who[i] = Who[i];
    __syncthreads();

    for (int t = 0; t < TSTEPS; ++t) {
      if (wv < 4) {
        const int Mt = wv >> 1, Ntl = wv & 1;
        floatx8 acc = {0, 0, 0, 0, 0, 0, 0, 0};
        const bf16* arow = th + (size_t)(Mt * 16 + l15) * NTOT + mod * NMOD;  // global A
        const bf16* brow = &sm.r.Wr[Ntl * 16 + l15][0];
        for (int kb = 0; kb < NMOD; kb += 32)
          acc = wmma_bf16(load_a16(arow, kb, halfL), load_b16(brow, kb, halfL), acc);
        const int rcol = Ntl * 16 + l15;
        #pragma unroll
        for (int q = 0; q < 8; ++q) {
          const int b = Mt * 16 + halfL * 8 + q;
          outYR[(((size_t)b * 32 + rcol) * 3 + mod) * TSTEPS + t] = tanhf(acc[q]);
        }
      } else if (mod == 2) {
        // y_t = tanh(th3 @ W_ho.T)
        const int id = tid - 128;
        const int b = id >> 2, g = id & 3;
        float s = 0.0f;
        const bf16* thr = th + (size_t)b * NTOT + 2 * NMOD + g * 256;
        for (int k = 0; k < 256; k += 8) {
          float4 raw = *(const float4*)(thr + k);
          const bf16* pv = (const bf16*)&raw;
          #pragma unroll
          for (int q = 0; q < 8; ++q) s += (float)pv[q] * sm.r.who[g * 256 + k + q];
        }
        sm.r.part[b][g] = s;
      }
      __syncthreads();
      if (mod == 2 && tid < 32) {
        const float s = sm.r.part[tid][0] + sm.r.part[tid][1] +
                        sm.r.part[tid][2] + sm.r.part[tid][3];
        outY[tid * TSTEPS + t] = tanhf(s);
      }
      grid_barrier(bcnt, bgen, NWG_TOTAL, lg);   // B1
      grid_barrier(bcnt, bgen, NWG_TOTAL, lg);   // B2
    }
  }
}

// ---------------- host launcher ----------------

extern "C" void kernel_launch(void* const* d_in, const int* in_sizes, int n_in,
                              void* d_out, int out_size, void* d_ws, size_t ws_size,
                              hipStream_t stream) {
  const float* x    = (const float*)d_in[0];
  const float* h0   = (const float*)d_in[1];
  const float* ctx  = (const float*)d_in[2];
  const float* iWN  = (const float*)d_in[3];
  const float* W_ih = (const float*)d_in[4];
  const float* W11  = (const float*)d_in[5];
  const float* W22  = (const float*)d_in[6];
  const float* W33  = (const float*)d_in[7];
  const float* W12  = (const float*)d_in[8];
  const float* W23  = (const float*)d_in[9];
  const float* Who  = (const float*)d_in[10];
  const float* WHc2 = (const float*)d_in[11];
  const float* Wr1  = (const float*)d_in[12];
  const float* Wr2  = (const float*)d_in[13];
  const float* Wr3  = (const float*)d_in[14];
  const int* idN12_1 = (const int*)d_in[15];
  const int* idN12_2 = (const int*)d_in[16];
  const int* idN23_2 = (const int*)d_in[17];
  const int* idN23_3 = (const int*)d_in[18];

  char* ws = (char*)d_ws;
  int*   bar   = (int*)(ws + 0);
  bf16*  th    = (bf16*)(ws + 256);
  float* wn    = (float*)(ws + 196864);
  bf16*  xbf   = (bf16*)(ws + 6488320);
  float* fm1   = (float*)(ws + 7536896);
  float* fm2   = (float*)(ws + 7563520);
  int*   inv12 = (int*)(ws + 7590144);
  int*   inv23 = (int*)(ws + 7594240);

  float* outH  = (float*)d_out;
  float* outY  = outH + (size_t)BATCH * NTOT * TSTEPS;
  float* outYR = outY + (size_t)BATCH * TSTEPS;

  setup_misc<<<1, 256, 0, stream>>>(h0, idN12_2, idN23_3, th, inv12, inv23, bar);
  setup_noise<<<(NTOT + 255) / 256, 256, 0, stream>>>(iWN, wn);
  setup_x<<<TSTEPS, 256, 0, stream>>>(x, xbf);
  rnn_persistent<<<NWG_TOTAL, 256, 0, stream>>>(
      h0, ctx, W_ih, W11, W22, W33, W12, W23, Who, WHc2, Wr1, Wr2, Wr3,
      idN12_1, idN23_2, th, wn, xbf, fm1, fm2, inv12, inv23, bar,
      outH, outY, outYR);
}
